// VariationalLSTM_84378927497403
// MI455X (gfx1250) — matrix operand order repaired and checked by reading, not compiled, not run on hardware
//
#include <hip/hip_runtime.h>

// Problem constants (match reference)
#define B_ 256
#define S_ 512
#define I_ 128
#define H_ 256

typedef __attribute__((ext_vector_type(16))) _Float16 v16h;
typedef __attribute__((ext_vector_type(8)))  _Float16 v8h;
typedef __attribute__((ext_vector_type(4)))  _Float16 v4h;
typedef __attribute__((ext_vector_type(8)))  float    v8f;

// sched_group_barrier masks
#define SGB_WMMA    0x008
#define SGB_VMEM_RD 0x020
#define SGB_DS_RD   0x100

static __device__ __forceinline__ v16h cat16(v8h lo, v8h hi) {
  return __builtin_shufflevector(lo, hi, 0,1,2,3,4,5,6,7,8,9,10,11,12,13,14,15);
}
// B fragment: K-halves of one weight row (lane L -> col N=L&15,
// lane half h takes K[8h:8h+8) and K[16+8h:24+8h) of the chunk).
static __device__ __forceinline__ v16h ldB(const _Float16* p) {
  return cat16(*(const v8h*)p, *(const v8h*)(p + 16));
}
static __device__ __forceinline__ float sigm(float x) {
  return 1.0f / (1.0f + __expf(-x));
}
static __device__ __forceinline__ float tanh_fast(float x) {
  return 2.0f / (1.0f + __expf(-2.0f * x)) - 1.0f;
}
static __device__ __forceinline__ v8f vzero() {
  v8f r;
#pragma unroll
  for (int i = 0; i < 8; ++i) r[i] = 0.0f;
  return r;
}

// ---------------------------------------------------------------------------
// Weight prep: f32 -> f16 ([N=4H][K] row-major == natural WMMA B layout),
// bias fusion b_ih + b_hh.
// ---------------------------------------------------------------------------
__global__ void cvt_f16_kernel(const float* __restrict__ s, _Float16* __restrict__ d, int n) {
  int i = blockIdx.x * blockDim.x + threadIdx.x;
  if (i < n) d[i] = (_Float16)s[i];
}
__global__ void bias_fuse_kernel(const float* __restrict__ a, const float* __restrict__ b,
                                 float* __restrict__ d, int n) {
  int i = blockIdx.x * blockDim.x + threadIdx.x;
  if (i < n) d[i] = a[i] + b[i];
}

// ---------------------------------------------------------------------------
// Input-projection GEMM (time-parallel, whole-chip):
//   Gx[bt*512 + t][wave][gate] = fragment of x_t @ Wih^T, f16, WMMA C layout
//   (lane L holds elements (M = e + (L>=16)*8, N = L&15), e = 0..7).
// grid = 16 batch-tiles x (512/TG) t-groups; TG timesteps share the B
// fragments in registers. Depth-1 sched choreography caps scheduler lookahead
// so accumulators + one prefetched chunk stay within the VGPR budget.
// ---------------------------------------------------------------------------
template <int KIN>
__global__ __launch_bounds__(512) void gate_gemm_kernel(
    const float* X,                      // [B, S, KIN]
    const _Float16* __restrict__ W,      // [4H, KIN] f16
    _Float16* __restrict__ Gx)           // [8192][16][4][256] f16
{
  constexpr int XCH = KIN / 32;
  constexpr int KV  = KIN / 4;
  constexpr int TG  = 2;

  __shared__ __align__(16) _Float16 sX[TG][16][KIN];

  const int tid  = threadIdx.x;
  const int wave = tid >> 5;
  const int lane = tid & 31;
  const int n    = lane & 15;
  const int hh   = lane >> 4;
  const int unit = wave * 16 + n;
  const int bt   = blockIdx.x >> 8;            // 0..15   (512/TG = 256 groups)
  const int t0   = (blockIdx.x & 255) * TG;    // 0..510
  const int klo  = hh * 8;

  size_t woff[4];
#pragma unroll
  for (int g = 0; g < 4; ++g) woff[g] = (size_t)(g * H_ + unit) * KIN + hh * 8;

  // Stage TG x-tiles (float4 load + packed f16 store).
  for (int e = tid; e < TG * 16 * KV; e += 512) {
    int j  = e / (16 * KV);
    int rr = (e / KV) & 15;
    int k4 = e % KV;
    const float4 f = *(const float4*)&X[((size_t)(bt * 16 + rr) * S_ + (t0 + j)) * KIN + k4 * 4];
    v4h p;
    p[0] = (_Float16)f.x; p[1] = (_Float16)f.y;
    p[2] = (_Float16)f.z; p[3] = (_Float16)f.w;
    *(v4h*)&sX[j][rr][k4 * 4] = p;
  }
  __syncthreads();

  v8f acc[TG][4];
#pragma unroll
  for (int c = 0; c < XCH; ++c) {
    const int kb = c * 32;
    v16h b0 = ldB(W + woff[0] + kb);
    v16h b1 = ldB(W + woff[1] + kb);
    v16h b2 = ldB(W + woff[2] + kb);
    v16h b3 = ldB(W + woff[3] + kb);
#pragma unroll
    for (int j = 0; j < TG; ++j) {
      v16h a = cat16(*(const v8h*)&sX[j][n][kb + klo],
                     *(const v8h*)&sX[j][n][kb + klo + 16]);
      if (c == 0) {
        acc[j][0] = __builtin_amdgcn_wmma_f32_16x16x32_f16(false, a, false, b0, (short)0, vzero(), false, false);
        acc[j][1] = __builtin_amdgcn_wmma_f32_16x16x32_f16(false, a, false, b1, (short)0, vzero(), false, false);
        acc[j][2] = __builtin_amdgcn_wmma_f32_16x16x32_f16(false, a, false, b2, (short)0, vzero(), false, false);
        acc[j][3] = __builtin_amdgcn_wmma_f32_16x16x32_f16(false, a, false, b3, (short)0, vzero(), false, false);
      } else {
        acc[j][0] = __builtin_amdgcn_wmma_f32_16x16x32_f16(false, a, false, b0, (short)0, acc[j][0], false, false);
        acc[j][1] = __builtin_amdgcn_wmma_f32_16x16x32_f16(false, a, false, b1, (short)0, acc[j][1], false, false);
        acc[j][2] = __builtin_amdgcn_wmma_f32_16x16x32_f16(false, a, false, b2, (short)0, acc[j][2], false, false);
        acc[j][3] = __builtin_amdgcn_wmma_f32_16x16x32_f16(false, a, false, b3, (short)0, acc[j][3], false, false);
      }
    }
  }

  // Depth-1 pipeline: prime chunk 0 loads; then [WMMAs of c][loads of c+1].
  __builtin_amdgcn_sched_group_barrier(SGB_DS_RD,   2 * TG, 0);
  __builtin_amdgcn_sched_group_barrier(SGB_VMEM_RD, 8,      0);
#pragma unroll
  for (int c = 0; c < XCH; ++c) {
    __builtin_amdgcn_sched_group_barrier(SGB_WMMA, 4 * TG, 0);
    if (c + 1 < XCH) {
      __builtin_amdgcn_sched_group_barrier(SGB_DS_RD,   2 * TG, 0);
      __builtin_amdgcn_sched_group_barrier(SGB_VMEM_RD, 8,      0);
    }
  }

  // Store fragments as f16 in WMMA C layout (1x b128 per lane per gate).
#pragma unroll
  for (int j = 0; j < TG; ++j) {
#pragma unroll
    for (int g = 0; g < 4; ++g) {
      v8h o;
#pragma unroll
      for (int i = 0; i < 8; ++i) o[i] = (_Float16)acc[j][g][i];
      *(v8h*)(Gx + ((((size_t)(bt * 512 + t0 + j)) * 16 + wave) * 4 + g) * 256 + lane * 8) = o;
    }
  }
}

// ---------------------------------------------------------------------------
// Recurrence with precomputed input projections: per step only h @ Whh^T
// (8 K-chunks). Gate fragments are dependence-free loads, prefetched at the
// region top; chunk pipeline depth-3 via sched_group_barrier.
// ---------------------------------------------------------------------------
template <bool LAST>
__global__ __launch_bounds__(512) void lstm_rec_pre_kernel(
    const _Float16* __restrict__ Gx,     // [8192][16][4][256] f16
    const _Float16* __restrict__ Whh,    // [4H, H] f16
    const float* __restrict__ bias,      // [4H] fused
    float* Y, float* outH, float* outC)
{
  constexpr int HCH = H_ / 32;
  constexpr int PF  = 3;
  constexpr size_t TS = (size_t)16 * 4 * 256;  // fragment halves per (bt, t)

  __shared__ __align__(16) _Float16 sH[16][H_];

  const int tid  = threadIdx.x;
  const int wave = tid >> 5;
  const int lane = tid & 31;
  const int n    = lane & 15;
  const int hh   = lane >> 4;
  const int unit = wave * 16 + n;
  const int row0 = blockIdx.x * 16;
  const int klo  = hh * 8;

  float bg[4];
#pragma unroll
  for (int g = 0; g < 4; ++g) bg[g] = bias[g * H_ + unit];

  size_t wohh[4];
#pragma unroll
  for (int g = 0; g < 4; ++g) wohh[g] = (size_t)(g * H_ + unit) * H_ + hh * 8;

  // Per-(bt, wave, lane) fragment base; advance by TS per timestep.
  const _Float16* gp =
      Gx + (((size_t)blockIdx.x * 512) * 16 + wave) * 4 * 256 + (size_t)lane * 8;

  v8f cst  = vzero();
  v8f hout = vzero();

  for (int e = tid; e < 16 * H_; e += 512) ((_Float16*)sH)[e] = (_Float16)0.0f;

  unsigned zoff = 0;  // laundered each step -> loop-carried

  for (int t = 0; t < S_; ++t) {
    asm volatile("" : "+s"(zoff));
    const _Float16* whh = Whh + zoff;

    __syncthreads();  // sH(t-1) writes visible; prior readers done

    // Precomputed gate fragments (f16 C layout) -> f32 accumulator init.
    v8h gxh[4];
#pragma unroll
    for (int g = 0; g < 4; ++g)
      gxh[g] = *(const v8h*)(gp + (size_t)t * TS + (size_t)g * 256);

    v8f acc[4];
#pragma unroll
    for (int c = 0; c < HCH; ++c) {
      const int kb = c * 32;
      v16h a  = cat16(*(const v8h*)&sH[n][kb + klo],
                      *(const v8h*)&sH[n][kb + klo + 16]);
      v16h b0 = ldB(whh + wohh[0] + kb);
      v16h b1 = ldB(whh + wohh[1] + kb);
      v16h b2 = ldB(whh + wohh[2] + kb);
      v16h b3 = ldB(whh + wohh[3] + kb);
      if (c == 0) {
        v8f c0, c1, c2, c3;
#pragma unroll
        for (int i = 0; i < 8; ++i) {
          c0[i] = (float)gxh[0][i]; c1[i] = (float)gxh[1][i];
          c2[i] = (float)gxh[2][i]; c3[i] = (float)gxh[3][i];
        }
        acc[0] = __builtin_amdgcn_wmma_f32_16x16x32_f16(false, a, false, b0, (short)0, c0, false, false);
        acc[1] = __builtin_amdgcn_wmma_f32_16x16x32_f16(false, a, false, b1, (short)0, c1, false, false);
        acc[2] = __builtin_amdgcn_wmma_f32_16x16x32_f16(false, a, false, b2, (short)0, c2, false, false);
        acc[3] = __builtin_amdgcn_wmma_f32_16x16x32_f16(false, a, false, b3, (short)0, c3, false, false);
      } else {
        acc[0] = __builtin_amdgcn_wmma_f32_16x16x32_f16(false, a, false, b0, (short)0, acc[0], false, false);
        acc[1] = __builtin_amdgcn_wmma_f32_16x16x32_f16(false, a, false, b1, (short)0, acc[1], false, false);
        acc[2] = __builtin_amdgcn_wmma_f32_16x16x32_f16(false, a, false, b2, (short)0, acc[2], false, false);
        acc[3] = __builtin_amdgcn_wmma_f32_16x16x32_f16(false, a, false, b3, (short)0, acc[3], false, false);
      }
    }

    // Schedule: gate fragments + PF chunks primed, then 4 WMMA / 10 loads.
    __builtin_amdgcn_sched_group_barrier(SGB_VMEM_RD, 4, 0);  // gx fragments
#pragma unroll
    for (int q = 0; q < PF; ++q) {
      __builtin_amdgcn_sched_group_barrier(SGB_DS_RD,   2, 0);
      __builtin_amdgcn_sched_group_barrier(SGB_VMEM_RD, 8, 0);
    }
#pragma unroll
    for (int c = 0; c < HCH; ++c) {
      __builtin_amdgcn_sched_group_barrier(SGB_WMMA, 4, 0);
      if (c + PF < HCH) {
        __builtin_amdgcn_sched_group_barrier(SGB_DS_RD,   2, 0);
        __builtin_amdgcn_sched_group_barrier(SGB_VMEM_RD, 8, 0);
      }
    }

#pragma unroll
    for (int e = 0; e < 8; ++e) {
      float iv = sigm(acc[0][e] + bg[0]);
      float fv = sigm(acc[1][e] + bg[1]);
      float gv = tanh_fast(acc[2][e] + bg[2]);
      float ov = sigm(acc[3][e] + bg[3]);
      float cv = fv * cst[e] + iv * gv;
      cst[e]  = cv;
      hout[e] = ov * tanh_fast(cv);
    }

    __syncthreads();  // all waves done reading sH(t-1)

#pragma unroll
    for (int e = 0; e < 8; ++e) {
      int m = e + hh * 8;
      sH[m][unit] = (_Float16)hout[e];
      Y[((size_t)(row0 + m) * S_ + t) * H_ + unit] = hout[e];
    }
  }

  if (LAST) {
#pragma unroll
    for (int e = 0; e < 8; ++e) {
      int m = e + hh * 8;
      outH[(size_t)(row0 + m) * H_ + unit] = hout[e];
      outC[(size_t)(row0 + m) * H_ + unit] = cst[e];
    }
  }
}

// ---------------------------------------------------------------------------
// Fallback fused layer kernel (used when d_ws is too small for the gate
// buffer). Identical to the round-4 kernel.
// ---------------------------------------------------------------------------
template <int KIN, bool LAST>
__global__ __launch_bounds__(512) void lstm_layer_kernel(
    const float* X, const _Float16* __restrict__ Wih,
    const _Float16* __restrict__ Whh, const float* __restrict__ bias,
    float* Y, float* outH, float* outC)
{
  constexpr int XCH = KIN / 32;
  constexpr int HCH = H_ / 32;
  constexpr int NCH = XCH + HCH;
  constexpr int KV  = KIN / 4;
  constexpr int PF  = 3;

  __shared__ __align__(16) _Float16 sX[16][KIN];
  __shared__ __align__(16) _Float16 sH[16][H_];

  const int tid  = threadIdx.x;
  const int wave = tid >> 5;
  const int lane = tid & 31;
  const int n    = lane & 15;
  const int hh   = lane >> 4;
  const int unit = wave * 16 + n;
  const int row0 = blockIdx.x * 16;
  const int klo  = hh * 8;

  float bg[4];
#pragma unroll
  for (int g = 0; g < 4; ++g) bg[g] = bias[g * H_ + unit];

  size_t woih[4], wohh[4];
#pragma unroll
  for (int g = 0; g < 4; ++g) {
    woih[g] = (size_t)(g * H_ + unit) * KIN + hh * 8;
    wohh[g] = (size_t)(g * H_ + unit) * H_  + hh * 8;
  }

  v8f cst  = vzero();
  v8f hout = vzero();

  for (int e = tid; e < 16 * H_; e += 512) ((_Float16*)sH)[e] = (_Float16)0.0f;
  __syncthreads();

  unsigned zoff = 0;

  for (int t = 0; t < S_; ++t) {
    asm volatile("" : "+s"(zoff));
    const _Float16* wih = Wih + zoff;
    const _Float16* whh = Whh + zoff;

    for (int e = tid; e < 16 * KV; e += 512) {
      int r = e / KV, k4 = e % KV;
      const float4 f = *(const float4*)&X[((size_t)(row0 + r) * S_ + t) * KIN + k4 * 4];
      v4h p;
      p[0] = (_Float16)f.x; p[1] = (_Float16)f.y;
      p[2] = (_Float16)f.z; p[3] = (_Float16)f.w;
      *(v4h*)&sX[r][k4 * 4] = p;
    }
    __syncthreads();

#define LD_A_(c)                                                              \
    ((c) < XCH                                                                \
         ? cat16(*(const v8h*)&sX[n][(c) * 32 + klo],                         \
                 *(const v8h*)&sX[n][(c) * 32 + klo + 16])                    \
         : cat16(*(const v8h*)&sH[n][((c) - XCH) * 32 + klo],                 \
                 *(const v8h*)&sH[n][((c) - XCH) * 32 + klo + 16]))
#define LD_B_(c, g)                                                           \
    ((c) < XCH ? ldB(wih + woih[g] + (c) * 32)                                \
               : ldB(whh + wohh[g] + ((c) - XCH) * 32))

    v8f acc[4];
#pragma unroll
    for (int c = 0; c < NCH; ++c) {
      v16h a  = LD_A_(c);
      v16h b0 = LD_B_(c, 0);
      v16h b1 = LD_B_(c, 1);
      v16h b2 = LD_B_(c, 2);
      v16h b3 = LD_B_(c, 3);
      if (c == 0) {
        acc[0] = __builtin_amdgcn_wmma_f32_16x16x32_f16(false, a, false, b0, (short)0, vzero(), false, false);
        acc[1] = __builtin_amdgcn_wmma_f32_16x16x32_f16(false, a, false, b1, (short)0, vzero(), false, false);
        acc[2] = __builtin_amdgcn_wmma_f32_16x16x32_f16(false, a, false, b2, (short)0, vzero(), false, false);
        acc[3] = __builtin_amdgcn_wmma_f32_16x16x32_f16(false, a, false, b3, (short)0, vzero(), false, false);
      } else {
        acc[0] = __builtin_amdgcn_wmma_f32_16x16x32_f16(false, a, false, b0, (short)0, acc[0], false, false);
        acc[1] = __builtin_amdgcn_wmma_f32_16x16x32_f16(false, a, false, b1, (short)0, acc[1], false, false);
        acc[2] = __builtin_amdgcn_wmma_f32_16x16x32_f16(false, a, false, b2, (short)0, acc[2], false, false);
        acc[3] = __builtin_amdgcn_wmma_f32_16x16x32_f16(false, a, false, b3, (short)0, acc[3], false, false);
      }
    }
#undef LD_A_
#undef LD_B_

#pragma unroll
    for (int q = 0; q < PF; ++q) {
      __builtin_amdgcn_sched_group_barrier(SGB_DS_RD,   2, 0);
      __builtin_amdgcn_sched_group_barrier(SGB_VMEM_RD, 8, 0);
    }
#pragma unroll
    for (int c = 0; c < NCH; ++c) {
      __builtin_amdgcn_sched_group_barrier(SGB_WMMA, 4, 0);
      if (c + PF < NCH) {
        __builtin_amdgcn_sched_group_barrier(SGB_DS_RD,   2, 0);
        __builtin_amdgcn_sched_group_barrier(SGB_VMEM_RD, 8, 0);
      }
    }

#pragma unroll
    for (int e = 0; e < 8; ++e) {
      float iv = sigm(acc[0][e] + bg[0]);
      float fv = sigm(acc[1][e] + bg[1]);
      float gv = tanh_fast(acc[2][e] + bg[2]);
      float ov = sigm(acc[3][e] + bg[3]);
      float cv = fv * cst[e] + iv * gv;
      cst[e]  = cv;
      hout[e] = ov * tanh_fast(cv);
    }

    __syncthreads();

#pragma unroll
    for (int e = 0; e < 8; ++e) {
      int m = e + hh * 8;
      sH[m][unit] = (_Float16)hout[e];
      Y[((size_t)(row0 + m) * S_ + t) * H_ + unit] = hout[e];
    }
  }

  if (LAST) {
#pragma unroll
    for (int e = 0; e < 8; ++e) {
      int m = e + hh * 8;
      outH[(size_t)(row0 + m) * H_ + unit] = hout[e];
      outC[(size_t)(row0 + m) * H_ + unit] = cst[e];
    }
  }
}

// ---------------------------------------------------------------------------
extern "C" void kernel_launch(void* const* d_in, const int* in_sizes, int n_in,
                              void* d_out, int out_size, void* d_ws, size_t ws_size,
                              hipStream_t stream) {
  const float* x    = (const float*)d_in[0];
  const float* Wih0 = (const float*)d_in[1];
  const float* Whh0 = (const float*)d_in[2];
  const float* bih0 = (const float*)d_in[3];
  const float* bhh0 = (const float*)d_in[4];
  const float* Wih1 = (const float*)d_in[5];
  const float* Whh1 = (const float*)d_in[6];
  const float* bih1 = (const float*)d_in[7];
  const float* bhh1 = (const float*)d_in[8];

  // ws layout: [0, 2MB) f16 weights + fused biases; [2MB, 2MB+256MB) gate buf.
  const size_t nW0ih = (size_t)4 * H_ * I_;  // 131072
  const size_t nWhh  = (size_t)4 * H_ * H_;  // 262144
  char* ws = (char*)d_ws;
  _Float16* w0ih = (_Float16*)ws;
  _Float16* w0hh = w0ih + nW0ih;
  _Float16* w1ih = w0hh + nWhh;
  _Float16* w1hh = w1ih + nWhh;
  float* bias0 = (float*)(w1hh + nWhh);
  float* bias1 = bias0 + 4 * H_;

  const size_t GOFF   = (size_t)2 << 20;                       // 2 MB
  const size_t gbytes = (size_t)16 * 512 * 16 * 4 * 256 * 2;   // 256 MB
  _Float16* gbuf = (_Float16*)(ws + GOFF);
  const bool pre = ws_size >= GOFF + gbytes;

  float* Y    = (float*)d_out;                       // [B,S,H] (y0 then y1 in place)
  float* outH = Y + (size_t)B_ * S_ * H_;            // [1,B,H]
  float* outC = outH + (size_t)B_ * H_;              // [1,B,H]

  // Weight prep
  cvt_f16_kernel<<<(int)((nW0ih + 255) / 256), 256, 0, stream>>>(Wih0, w0ih, (int)nW0ih);
  cvt_f16_kernel<<<(int)((nWhh  + 255) / 256), 256, 0, stream>>>(Whh0, w0hh, (int)nWhh);
  cvt_f16_kernel<<<(int)((nWhh  + 255) / 256), 256, 0, stream>>>(Wih1, w1ih, (int)nWhh);
  cvt_f16_kernel<<<(int)((nWhh  + 255) / 256), 256, 0, stream>>>(Whh1, w1hh, (int)nWhh);
  bias_fuse_kernel<<<4, 256, 0, stream>>>(bih0, bhh0, bias0, 4 * H_);
  bias_fuse_kernel<<<4, 256, 0, stream>>>(bih1, bhh1, bias1, 4 * H_);

  if (pre) {
    // Layer 1: time-parallel input GEMM, then h-only recurrence.
    gate_gemm_kernel<I_><<<16 * 256, 512, 0, stream>>>(x, w0ih, gbuf);
    lstm_rec_pre_kernel<false><<<B_ / 16, 512, 0, stream>>>(
        gbuf, w0hh, bias0, Y, nullptr, nullptr);
    // Layer 2: same, input is y0 (read fully by the GEMM before rec overwrites).
    gate_gemm_kernel<H_><<<16 * 256, 512, 0, stream>>>(Y, w1ih, gbuf);
    lstm_rec_pre_kernel<true><<<B_ / 16, 512, 0, stream>>>(
        gbuf, w1hh, bias1, Y, outH, outC);
  } else {
    // Fallback: fused kernels (x + h chunks on the recurrent path).
    lstm_layer_kernel<I_, false><<<B_ / 16, 512, 0, stream>>>(
        x, w0ih, w0hh, bias0, Y, nullptr, nullptr);
    lstm_layer_kernel<H_, true><<<B_ / 16, 512, 0, stream>>>(
        Y, w1ih, w1hh, bias1, Y, outH, outC);
  }
}
